// AttentionLayer_23407571763526
// MI455X (gfx1250) — matrix operand order repaired and checked by reading, not compiled, not run on hardware
//
#include <hip/hip_runtime.h>
#include <hip/hip_bf16.h>

// ---------------------------------------------------------------------------
// AttentionLayer for MI455X (gfx1250): f16 WMMA (f32 accumulate) everywhere.
// D=1024, H=16, HD=64, B=8, S=1024, PAD=1
// GEMMs use LDS-staged, double-buffered B tiles via the CDNA5 async-to-LDS
// path (ASYNCcnt) + register-double-buffered A fragments for latency hiding.
// ---------------------------------------------------------------------------

typedef __attribute__((ext_vector_type(16))) _Float16 v16h;
typedef __attribute__((ext_vector_type(8)))  _Float16 v8h;
typedef __attribute__((ext_vector_type(8)))  float    v8f;
typedef int v4i_vs __attribute__((vector_size(16)));   // matches builtin param

#define BATCH 8
#define SEQ   1024
#define DIM   1024
#define NH    16
#define HD    64
#define PADID 1

#if __has_builtin(__builtin_amdgcn_global_load_async_to_lds_b128)
#define USE_ASYNC_LDS 1
#else
#define USE_ASYNC_LDS 0
#endif

__device__ __forceinline__ v8f wmma_f16(v16h a, v16h b, v8f c) {
  // D = A(16x32) * B(32x16) + C, f32 accumulate
  return __builtin_amdgcn_wmma_f32_16x16x32_f16(
      /*neg_a=*/false, a, /*neg_b=*/false, b,
      /*c_mod=*/(short)0, c, /*reuse_a=*/false, /*reuse_b=*/false);
}

// Move 16 bytes global -> LDS. Prefer the gfx1250 async-tensor path (ASYNCcnt).
__device__ __forceinline__ void stage_b128(const _Float16* g, _Float16* l) {
#if USE_ASYNC_LDS
  __builtin_amdgcn_global_load_async_to_lds_b128(
      (__attribute__((address_space(1))) v4i_vs*)(void*)g,
      (__attribute__((address_space(3))) v4i_vs*)(void*)l, 0, 0);
#else
  *(v8h*)l = *(const v8h*)g;
#endif
}

__device__ __forceinline__ void async_wait_all() {
#if USE_ASYNC_LDS
#if __has_builtin(__builtin_amdgcn_s_wait_asynccnt)
  __builtin_amdgcn_s_wait_asynccnt(0);
#else
  asm volatile("s_wait_asynccnt 0x0" ::: "memory");
#endif
#endif
}

// A fragment: 16x32 f16, rows m over contraction K. Memory row-major, ld elems.
// Lane l (<16): row m=l, elems 0..7 = K 0..7, elems 8..15 = K 16..23
// Lane l (>=16): row m=l-16, elems 0..7 = K 8..15, elems 8..15 = K 24..31
__device__ __forceinline__ v16h load_a16x32(const _Float16* base, int ld) {
  const int lane = threadIdx.x & 31;
  const int m    = lane & 15;
  const int kLo  = (lane & 16) ? 8 : 0;
  const _Float16* p = base + (size_t)m * ld + kLo;
  v8h lo = *(const v8h*)(p);
  v8h hi = *(const v8h*)(p + 16);
  v16h f;
#pragma unroll
  for (int i = 0; i < 8; ++i) { f[i] = lo[i]; f[i + 8] = hi[i]; }
  return f;
}

// B fragment: 32x16 f16, element (k,n) = Brow[n*ld + k]  (computes A @ Brow^T).
// Lane l (<16): column n=l, K 0..7,16..23 ; lane l (>=16): column n=l-16, K 8..15,24..31.
__device__ __forceinline__ v16h load_bT16x32(const _Float16* base, int ld) {
  const int lane = threadIdx.x & 31;
  const int n    = lane & 15;
  const int kOff = (lane & 16) ? 8 : 0;
  const _Float16* p = base + (size_t)n * ld + kOff;
  v8h lo = *(const v8h*)(p);
  v8h hi = *(const v8h*)(p + 16);
  v16h f;
#pragma unroll
  for (int i = 0; i < 8; ++i) { f[i] = lo[i]; f[i + 8] = hi[i]; }
  return f;
}

// ---------------------------------------------------------------------------
__global__ void cvt_f32_f16(const float* __restrict__ src,
                            _Float16* __restrict__ dst, int n) {
  int i = blockIdx.x * blockDim.x + threadIdx.x;
  if (i < n) dst[i] = (_Float16)src[i];
}

// ---------------------------------------------------------------------------
// Core pipelined GEMM tile: block = 256 thr (8 waves), block tile M=128,N=64.
// A: [M,1024] f16 row-major (per-wave, register double-buffered from global)
// W: rows are output columns, [*,1024] f16 row-major (LDS double-buffered)
// acc: per-wave 16x64 tile as 4 x v8f.
__device__ __forceinline__ void gemm_tile_128x64(
    const _Float16* __restrict__ Arow,   // + m0*DIM already applied
    const _Float16* __restrict__ Wrow,   // + ncol*DIM already applied
    _Float16 (*Bs)[64 * 32],             // [2][64*32] shared
    v8f acc[4]) {
  const int tid  = threadIdx.x;
  const int brow = tid >> 2;             // 0..63 : output column within tile
  const int bseg = (tid & 3) * 8;        // 0,8,16,24 : K segment (halves)

  // prime buffer 0 with k=0
  stage_b128(Wrow + (size_t)brow * DIM + bseg, &Bs[0][brow * 32 + bseg]);
  v16h aCur = load_a16x32(Arow, DIM);

  for (int kt = 0; kt < 32; ++kt) {
    const int buf = kt & 1;
    async_wait_all();          // my wave's async stores to LDS are done
    __syncthreads();           // everyone's are done; prior reads drained

    if (kt + 1 < 32) {         // kick next tile while computing this one
      const int kn = (kt + 1) * 32;
      stage_b128(Wrow + (size_t)brow * DIM + kn + bseg,
                 &Bs[buf ^ 1][brow * 32 + bseg]);
    }
    v16h aNext = aCur;
    if (kt + 1 < 32) aNext = load_a16x32(Arow + (kt + 1) * 32, DIM);

    v16h b0 = load_bT16x32(&Bs[buf][0 * 16 * 32], 32);
    v16h b1 = load_bT16x32(&Bs[buf][1 * 16 * 32], 32);
    v16h b2 = load_bT16x32(&Bs[buf][2 * 16 * 32], 32);
    v16h b3 = load_bT16x32(&Bs[buf][3 * 16 * 32], 32);
    acc[0] = wmma_f16(aCur, b0, acc[0]);
    acc[1] = wmma_f16(aCur, b1, acc[1]);
    acc[2] = wmma_f16(aCur, b2, acc[2]);
    acc[3] = wmma_f16(aCur, b3, acc[3]);
    aCur = aNext;
  }
}

// ---------------------------------------------------------------------------
// Fused QKV projection: out(m,n) = sum_k x(m,k) * W(n,k) + b(n)
// M = 8192, N' = 3072, K = 1024. Grid (48,64), block 256.
// Q scaled by 1/sqrt(HD); Q,K stored [B,H,S,HD]; V stored transposed [B,H,HD,S].
__global__ __launch_bounds__(256)
void qkv_gemm_kernel(const _Float16* __restrict__ xh,
                     const _Float16* __restrict__ wqh,
                     const _Float16* __restrict__ wkh,
                     const _Float16* __restrict__ wvh,
                     const float* __restrict__ bq,
                     const float* __restrict__ bk,
                     const float* __restrict__ bv,
                     _Float16* __restrict__ Qd,
                     _Float16* __restrict__ Kd,
                     _Float16* __restrict__ Vt) {
  __shared__ _Float16 Bs[2][64 * 32];
  const int lane = threadIdx.x & 31;
  const int wave = threadIdx.x >> 5;
  const int n0   = blockIdx.x * 64;                 // 0..3071
  const int m0   = blockIdx.y * 128 + wave * 16;    // 0..8176
  const int which = n0 >> 10;                       // 0=Q 1=K 2=V (block-uniform)
  const int ncol  = n0 & 1023;
  const _Float16* W   = (which == 0) ? wqh : (which == 1) ? wkh : wvh;
  const float*    bia = (which == 0) ? bq  : (which == 1) ? bk  : bv;

  v8f acc[4] = {};
  gemm_tile_128x64(xh + (size_t)m0 * DIM, W + (size_t)ncol * DIM, Bs, acc);

  // Epilogue: C-layout scatter. Lane covers column n, rows mh..mh+7.
  const int n     = lane & 15;
  const int mh    = (lane >> 4) * 8;
  const int bIdx  = m0 >> 10;
  const int sBase = (m0 & 1023) + mh;
#pragma unroll
  for (int t = 0; t < 4; ++t) {
    const int   col = ncol + t * 16 + n;
    const float bb  = bia[col];
    const int   h   = col >> 6;
    const int   hd  = col & 63;
    const size_t bh = (size_t)bIdx * NH + h;
#pragma unroll
    for (int r = 0; r < 8; ++r) {
      const float val = acc[t][r] + bb;
      const int   s   = sBase + r;
      if (which == 0)
        Qd[(bh * SEQ + s) * HD + hd] = (_Float16)(val * 0.125f);  // 1/sqrt(64)
      else if (which == 1)
        Kd[(bh * SEQ + s) * HD + hd] = (_Float16)val;
      else
        Vt[(bh * HD + hd) * SEQ + s] = (_Float16)val;
    }
  }
}

// ---------------------------------------------------------------------------
// Flash attention: grid (B*H, S/64), block 128 (4 waves, 16 queries each).
__global__ __launch_bounds__(128)
void attn_kernel(const _Float16* __restrict__ Qd,
                 const _Float16* __restrict__ Kd,
                 const _Float16* __restrict__ Vt,
                 const int* __restrict__ ids,      // [B,S]
                 _Float16* __restrict__ Oh) {      // [B,S,D]
  __shared__ _Float16 pbuf[4][16 * 32];            // per-wave P tile (16q x 32k)
  const int lane = threadIdx.x & 31;
  const int wave = threadIdx.x >> 5;
  const int bh   = blockIdx.x;                     // b*16 + h
  const int qblk = blockIdx.y;                     // 0..15
  const int b    = bh >> 4;
  const int q0   = qblk * 64 + wave * 16;

  const _Float16* Qbh = Qd + (size_t)bh * SEQ * HD;
  const _Float16* Kbh = Kd + (size_t)bh * SEQ * HD;
  const _Float16* Vbh = Vt + (size_t)bh * HD * SEQ;

  const v16h aQ0 = load_a16x32(Qbh + (size_t)q0 * HD + 0,  HD);
  const v16h aQ1 = load_a16x32(Qbh + (size_t)q0 * HD + 32, HD);

  v8f o[4] = {};
  float mrow[8], lrow[8];
#pragma unroll
  for (int r = 0; r < 8; ++r) { mrow[r] = -1e30f; lrow[r] = 0.0f; }

  const int n  = lane & 15;
  const int mh = (lane >> 4) * 8;
  const int nKB = (qblk + 1) * 2;                  // uniform over the block

  for (int kb = 0; kb < nKB; ++kb) {
    const int kBase = kb * 32;

    // ---- scores: 16q x 32k = two 16x16 tiles, each K=64 contraction ----
    v16h bk00 = load_bT16x32(Kbh + (size_t)kBase * HD + 0,  HD);
    v16h bk01 = load_bT16x32(Kbh + (size_t)kBase * HD + 32, HD);
    v16h bk10 = load_bT16x32(Kbh + (size_t)(kBase + 16) * HD + 0,  HD);
    v16h bk11 = load_bT16x32(Kbh + (size_t)(kBase + 16) * HD + 32, HD);
    v8f s0 = {}, s1 = {};
    s0 = wmma_f16(aQ0, bk00, s0);
    s0 = wmma_f16(aQ1, bk01, s0);
    s1 = wmma_f16(aQ0, bk10, s1);
    s1 = wmma_f16(aQ1, bk11, s1);

    const int  key0 = kBase + n;
    const int  key1 = key0 + 16;
    const bool ok0  = (ids[b * SEQ + key0] != PADID);
    const bool ok1  = (ids[b * SEQ + key1] != PADID);

    // ---- mask + online softmax (rows live across 16 lanes per half-wave) ----
    _Float16 ph0[8], ph1[8];
#pragma unroll
    for (int r = 0; r < 8; ++r) {
      const int q = q0 + mh + r;
      float v0 = (ok0 && key0 <= q) ? s0[r] : -1e30f;
      float v1 = (ok1 && key1 <= q) ? s1[r] : -1e30f;
      float rmax = fmaxf(v0, v1);
      rmax = fmaxf(rmax, __shfl_xor(rmax, 8, 32));
      rmax = fmaxf(rmax, __shfl_xor(rmax, 4, 32));
      rmax = fmaxf(rmax, __shfl_xor(rmax, 2, 32));
      rmax = fmaxf(rmax, __shfl_xor(rmax, 1, 32));
      const float mnew  = fmaxf(mrow[r], rmax);
      const float alpha = __expf(mrow[r] - mnew);
      const float p0 = __expf(v0 - mnew);
      const float p1 = __expf(v1 - mnew);
      float rs = p0 + p1;
      rs += __shfl_xor(rs, 8, 32);
      rs += __shfl_xor(rs, 4, 32);
      rs += __shfl_xor(rs, 2, 32);
      rs += __shfl_xor(rs, 1, 32);
      lrow[r] = lrow[r] * alpha + rs;
      mrow[r] = mnew;
#pragma unroll
      for (int t = 0; t < 4; ++t) o[t][r] *= alpha;
      ph0[r] = (_Float16)p0;
      ph1[r] = (_Float16)p1;
    }

    // ---- re-layout P (C-layout -> A-layout) through LDS ----
    __syncthreads();
    _Float16* pb = pbuf[wave];
#pragma unroll
    for (int r = 0; r < 8; ++r) {
      pb[(mh + r) * 32 + n]      = ph0[r];
      pb[(mh + r) * 32 + n + 16] = ph1[r];
    }
    __syncthreads();
    const v16h aP = load_a16x32(pb, 32);

    // ---- O += P @ V : 4 hd-tiles of 16 columns ----
    v16h bv0 = load_bT16x32(Vbh + (size_t)(0 * 16) * SEQ + kBase, SEQ);
    v16h bv1 = load_bT16x32(Vbh + (size_t)(1 * 16) * SEQ + kBase, SEQ);
    v16h bv2 = load_bT16x32(Vbh + (size_t)(2 * 16) * SEQ + kBase, SEQ);
    v16h bv3 = load_bT16x32(Vbh + (size_t)(3 * 16) * SEQ + kBase, SEQ);
    o[0] = wmma_f16(aP, bv0, o[0]);
    o[1] = wmma_f16(aP, bv1, o[1]);
    o[2] = wmma_f16(aP, bv2, o[2]);
    o[3] = wmma_f16(aP, bv3, o[3]);
  }

  // ---- normalize + store attn output as [B,S,D] (heads concatenated) ----
  const int h = bh & 15;
#pragma unroll
  for (int t = 0; t < 4; ++t) {
#pragma unroll
    for (int r = 0; r < 8; ++r) {
      const int s   = q0 + mh + r;
      const int col = h * HD + t * 16 + n;
      Oh[((size_t)b * SEQ + s) * DIM + col] = (_Float16)(o[t][r] / lrow[r]);
    }
  }
}

// ---------------------------------------------------------------------------
// Output projection: out(m,n) = sum_k A(m,k) * Wo(n,k) + bo(n), f32 output.
// Grid (16,64), block 256.
__global__ __launch_bounds__(256)
void out_gemm_kernel(const _Float16* __restrict__ Ah,
                     const _Float16* __restrict__ woh,
                     const float* __restrict__ bo,
                     float* __restrict__ out) {
  __shared__ _Float16 Bs[2][64 * 32];
  const int lane = threadIdx.x & 31;
  const int wave = threadIdx.x >> 5;
  const int n0   = blockIdx.x * 64;
  const int m0   = blockIdx.y * 128 + wave * 16;

  v8f acc[4] = {};
  gemm_tile_128x64(Ah + (size_t)m0 * DIM, woh + (size_t)n0 * DIM, Bs, acc);

  const int n  = lane & 15;
  const int mh = (lane >> 4) * 8;
#pragma unroll
  for (int t = 0; t < 4; ++t) {
    const int   col = n0 + t * 16 + n;
    const float bb  = bo[col];
#pragma unroll
    for (int r = 0; r < 8; ++r)
      out[(size_t)(m0 + mh + r) * DIM + col] = acc[t][r] + bb;
  }
}

// ---------------------------------------------------------------------------
extern "C" void kernel_launch(void* const* d_in, const int* in_sizes, int n_in,
                              void* d_out, int out_size, void* d_ws, size_t ws_size,
                              hipStream_t stream) {
  const float* x   = (const float*)d_in[0];
  const int*   ids = (const int*)d_in[1];
  const float* Wq  = (const float*)d_in[2];
  const float* bq  = (const float*)d_in[3];
  const float* Wk  = (const float*)d_in[4];
  const float* bk  = (const float*)d_in[5];
  const float* Wv  = (const float*)d_in[6];
  const float* bv  = (const float*)d_in[7];
  const float* Wo  = (const float*)d_in[8];
  const float* bo  = (const float*)d_in[9];
  float* out = (float*)d_out;

  const size_t MB = 1024ull * 1024ull;
  char* ws = (char*)d_ws;
  _Float16* xh  = (_Float16*)(ws + 0);         // 16 MB  [8192,1024]
  _Float16* wqh = (_Float16*)(ws + 16 * MB);   //  2 MB
  _Float16* wkh = (_Float16*)(ws + 18 * MB);   //  2 MB
  _Float16* wvh = (_Float16*)(ws + 20 * MB);   //  2 MB
  _Float16* woh = (_Float16*)(ws + 22 * MB);   //  2 MB
  _Float16* Qd  = (_Float16*)(ws + 24 * MB);   // 16 MB  [B,H,S,HD] (pre-scaled)
  _Float16* Kd  = (_Float16*)(ws + 40 * MB);   // 16 MB  [B,H,S,HD]
  _Float16* Vt  = (_Float16*)(ws + 56 * MB);   // 16 MB  [B,H,HD,S]
  _Float16* Oh  = (_Float16*)(ws + 72 * MB);   // 16 MB  [B,S,D]

  const int nx = BATCH * SEQ * DIM;   // 8388608
  const int nw = DIM * DIM;           // 1048576
  cvt_f32_f16<<<nx / 256, 256, 0, stream>>>(x,  xh,  nx);
  cvt_f32_f16<<<nw / 256, 256, 0, stream>>>(Wq, wqh, nw);
  cvt_f32_f16<<<nw / 256, 256, 0, stream>>>(Wk, wkh, nw);
  cvt_f32_f16<<<nw / 256, 256, 0, stream>>>(Wv, wvh, nw);
  cvt_f32_f16<<<nw / 256, 256, 0, stream>>>(Wo, woh, nw);

  qkv_gemm_kernel<<<dim3(48, 64), 256, 0, stream>>>(
      xh, wqh, wkh, wvh, bq, bk, bv, Qd, Kd, Vt);

  attn_kernel<<<dim3(BATCH * NH, SEQ / 64), 128, 0, stream>>>(
      Qd, Kd, Vt, ids, Oh);

  out_gemm_kernel<<<dim3(16, 64), 256, 0, stream>>>(Oh, woh, bo, out);
}
